// LayerWeighterModel_73023033967043
// MI455X (gfx1250) — matrix operand order repaired and checked by reading, not compiled, not run on hardware
//
#include <hip/hip_runtime.h>
#include <hip/hip_bf16.h>

// Problem constants (from reference)
#define QN     4096
#define DN     32768
#define DIMN   128
#define NSLICE 16
#define DPS    (DN / NSLICE)    // 2048 docs per slice
#define DSTEP  32               // docs staged into LDS per step
#define NSTEPS (DPS / DSTEP)    // 64 steps per slice
#define LROW   136              // padded LDS row stride in ushorts (272 B = 68 dwords)

typedef __attribute__((ext_vector_type(16))) __bf16 v16bf;
typedef __attribute__((ext_vector_type(8)))  float  v8f;

struct __align__(16) U4 { unsigned int v[4]; };
struct U4x2 { U4 a, b; };

// ---------------------------------------------------------------------------
// fp32 -> bf16 (round-to-nearest-even)
static __device__ inline unsigned short f32_to_bf16_rne(float x) {
  unsigned int u = __float_as_uint(x);
  unsigned int r = u + 0x7FFFu + ((u >> 16) & 1u);
  return (unsigned short)(r >> 16);
}

// Kernel 1: split fp32 embeddings into bf16 hi/lo pairs (bf16x2 decomposition)
__global__ __launch_bounds__(256)
void convert_split_bf16(const float* __restrict__ qe,
                        const float* __restrict__ de,
                        unsigned short* __restrict__ qhi,
                        unsigned short* __restrict__ qlo,
                        unsigned short* __restrict__ dhi,
                        unsigned short* __restrict__ dlo) {
  const int QE = QN * DIMN;
  const int DE = DN * DIMN;
  int i = blockIdx.x * blockDim.x + threadIdx.x;
  if (i < QE) {
    float x = qe[i];
    unsigned short h = f32_to_bf16_rne(x);
    float hf = __uint_as_float(((unsigned int)h) << 16);
    qhi[i] = h;
    qlo[i] = f32_to_bf16_rne(x - hf);
  } else if (i < QE + DE) {
    int j = i - QE;
    float x = de[j];
    unsigned short h = f32_to_bf16_rne(x);
    float hf = __uint_as_float(((unsigned int)h) << 16);
    dhi[j] = h;
    dlo[j] = f32_to_bf16_rne(x - hf);
  }
}

// ---------------------------------------------------------------------------
// Build a 16-element bf16 fragment from two 16-byte chunks
static __device__ inline v16bf frag2(const unsigned short* p0,
                                     const unsigned short* p1) {
  U4x2 t;
  t.a = *reinterpret_cast<const U4*>(p0);
  t.b = *reinterpret_cast<const U4*>(p1);
  return __builtin_bit_cast(v16bf, t);
}

// ---------------------------------------------------------------------------
// Kernel 2: LDS-shared, double-buffered MaxSim GEMM.
// Block = 8 waves = 8 q-tiles (128 rows) x one 2048-doc slice.
// Doc tiles staged into LDS with global_load_async_to_lds_b128 (ASYNCcnt),
// shared by all 8 waves; compute = v_wmma_f32_16x16x32_bf16 (bf16x3 ~ fp32).
__global__ __launch_bounds__(256)
void maxsim_wmma(const unsigned short* __restrict__ qhi,
                 const unsigned short* __restrict__ qlo,
                 const unsigned short* __restrict__ dhi,
                 const unsigned short* __restrict__ dlo,
                 float* __restrict__ pmax,
                 int*   __restrict__ pidx) {
  // 2 buffers x 64 padded rows (32 hi docs, 32 lo docs) x 272 B  = 34,816 B
  __shared__ __align__(16) unsigned short lds[2][2 * DSTEP][LROW];

  const int t     = (int)threadIdx.x;
  const int lane  = t & 31;
  const int wave  = t >> 5;
  const int qgrp  = (int)blockIdx.x >> 4;   // 0..31
  const int slice = (int)blockIdx.x & 15;   // 0..15
  const int qtile = qgrp * 8 + wave;        // 0..255
  const int q0    = qtile * 16;
  const int dbase = slice * DPS;
  const int r15   = lane & 15;
  const int half  = lane >> 4;

  // ---- A fragments (query tile, hi/lo), register-resident for whole slice.
  // 16-bit A 16x32 layout: elements 0-7 at K = 32c + 8*half,
  //                        elements 8-15 at K = 32c + 16 + 8*half.
  const unsigned short* arh = qhi + (size_t)(q0 + r15) * DIMN;
  const unsigned short* arl = qlo + (size_t)(q0 + r15) * DIMN;
  v16bf Ahi[4], Alo[4];
#pragma unroll
  for (int c = 0; c < 4; ++c) {
    const int k = 32 * c;
    Ahi[c] = frag2(arh + k + 8 * half, arh + k + 16 + 8 * half);
    Alo[c] = frag2(arl + k + 8 * half, arl + k + 16 + 8 * half);
  }

  float rmax[8];
  int   ridx[8];
#pragma unroll
  for (int r = 0; r < 8; ++r) { rmax[r] = -3.402823466e38f; ridx[r] = 0; }

  // ---- async staging of one 32-doc tile (hi+lo) into LDS buffer `buf`.
  // 1024 16-byte vectors per tile; thread handles 4 of them.
  auto stage = [&](int buf, int d0) {
#pragma unroll
    for (int i = 0; i < 4; ++i) {
      const int v     = i * 256 + t;     // 0..1023
      const int row   = v >> 4;          // 0..63 (0-31 = hi, 32-63 = lo)
      const int chunk = v & 15;          // 16B chunk within 256B row
      const unsigned short* src =
          (i < 2) ? (dhi + (size_t)(d0 + row) * DIMN + chunk * 8)
                  : (dlo + (size_t)(d0 + row - DSTEP) * DIMN + chunk * 8);
      unsigned int ldsoff = (unsigned int)(size_t)&lds[buf][row][chunk * 8];
      asm volatile("global_load_async_to_lds_b128 %0, %1, off"
                   :: "v"(ldsoff), "v"(src) : "memory");
    }
  };

  stage(0, dbase);  // prologue: fill buffer 0 (ASYNCcnt = 4)

  for (int step = 0; step < NSTEPS; ++step) {
    const int cb = step & 1;
    // prefetch next tile into the other buffer (wraps harmlessly on last step)
    int nstep = step + 1; if (nstep == NSTEPS) nstep = 0;
    stage(cb ^ 1, dbase + nstep * DSTEP);
    // wait for current buffer's 4 async loads (in-order completion),
    // leaving the 4 just-issued prefetches in flight
    asm volatile("s_wait_asynccnt 0x4" ::: "memory");
    __syncthreads();

    const int d0 = dbase + step * DSTEP;
#pragma unroll
    for (int s = 0; s < 2; ++s) {          // two 16-doc sub-tiles
      v8f acc = {0.f, 0.f, 0.f, 0.f, 0.f, 0.f, 0.f, 0.f};
#pragma unroll
      for (int c = 0; c < 4; ++c) {
        // 16-bit B 32x16 layout: elements 0-7 at K = 32c + 8*half, 8-15 at K = 32c + 16 + 8*half
        const int k = 32 * c + 8 * half;
        v16bf Bh = frag2(&lds[cb][16 * s + r15][k],
                         &lds[cb][16 * s + r15][k + 16]);
        v16bf Bl = frag2(&lds[cb][DSTEP + 16 * s + r15][k],
                         &lds[cb][DSTEP + 16 * s + r15][k + 16]);
        // sim ~= Ahi*Bh + Ahi*Bl + Alo*Bh   (bf16x3 ~ fp32 accuracy)
        acc = __builtin_amdgcn_wmma_f32_16x16x32_bf16(false, Ahi[c], false, Bh,
                                                      (short)0, acc, false, false);
        acc = __builtin_amdgcn_wmma_f32_16x16x32_bf16(false, Ahi[c], false, Bl,
                                                      (short)0, acc, false, false);
        acc = __builtin_amdgcn_wmma_f32_16x16x32_bf16(false, Alo[c], false, Bh,
                                                      (short)0, acc, false, false);
      }
      // C layout: VGPR r, lanes 0-15 -> row r, col=lane; lanes 16-31 -> row r+8
      const int col = d0 + 16 * s + r15;
#pragma unroll
      for (int r = 0; r < 8; ++r) {
        if (acc[r] > rmax[r]) { rmax[r] = acc[r]; ridx[r] = col; }
      }
    }
    __syncthreads();  // compute done before next iteration overwrites cb
  }

  // Reduce (max, argmax) across the 16 lanes of each half-wave
#pragma unroll
  for (int m = 1; m < 16; m <<= 1) {
#pragma unroll
    for (int r = 0; r < 8; ++r) {
      float om = __shfl_xor(rmax[r], m, 32);
      int   oi = __shfl_xor(ridx[r], m, 32);
      if (om > rmax[r] || (om == rmax[r] && oi < ridx[r])) {
        rmax[r] = om; ridx[r] = oi;
      }
    }
  }

  if (r15 == 0) {
    const int rowbase = q0 + 8 * half;
#pragma unroll
    for (int r = 0; r < 8; ++r) {
      pmax[(size_t)(rowbase + r) * NSLICE + slice] = rmax[r];
      pidx[(size_t)(rowbase + r) * NSLICE + slice] = ridx[r];
    }
  }
}

// ---------------------------------------------------------------------------
// Kernel 3: reduce slice partials per query, gather token weights,
// normalize (with zero-sum uniform fallback), emit scalar.
__global__ __launch_bounds__(256)
void finalize_maxsim(const float* __restrict__ pmax, const int* __restrict__ pidx,
                     const int* __restrict__ qids, const int* __restrict__ dids,
                     const float* __restrict__ qwt, const float* __restrict__ dwt,
                     float* __restrict__ out) {
  __shared__ float s_dot[256];
  __shared__ float s_comb[256];
  __shared__ float s_mx[256];

  float dot = 0.f, comb = 0.f, mxsum = 0.f;
  for (int q = (int)threadIdx.x; q < QN; q += 256) {
    float best = -3.402823466e38f;
    int   bidx = 0;
    for (int s = 0; s < NSLICE; ++s) {
      float v = pmax[(size_t)q * NSLICE + s];
      int   i = pidx[(size_t)q * NSLICE + s];
      if (v > best || (v == best && i < bidx)) { best = v; bidx = i; }
    }
    float qw = qwt[qids[q]];
    float dw = dwt[dids[bidx]];
    float c  = qw * dw;
    comb  += c;
    dot   += best * c;
    mxsum += best;
  }
  s_dot[threadIdx.x]  = dot;
  s_comb[threadIdx.x] = comb;
  s_mx[threadIdx.x]   = mxsum;
  __syncthreads();
  for (int off = 128; off > 0; off >>= 1) {
    if ((int)threadIdx.x < off) {
      s_dot[threadIdx.x]  += s_dot[threadIdx.x + off];
      s_comb[threadIdx.x] += s_comb[threadIdx.x + off];
      s_mx[threadIdx.x]   += s_mx[threadIdx.x + off];
    }
    __syncthreads();
  }
  if (threadIdx.x == 0) {
    float wsum = s_comb[0];
    out[0] = (wsum > 0.f) ? (s_dot[0] / wsum) : (s_mx[0] / (float)QN);
  }
}

// ---------------------------------------------------------------------------
extern "C" void kernel_launch(void* const* d_in, const int* in_sizes, int n_in,
                              void* d_out, int out_size, void* d_ws, size_t ws_size,
                              hipStream_t stream) {
  (void)in_sizes; (void)n_in; (void)out_size; (void)ws_size;

  const float* qe   = (const float*)d_in[0];  // [Q,128]
  const float* de   = (const float*)d_in[1];  // [D,128]
  const int*   qids = (const int*)d_in[2];    // [Q]
  const int*   dids = (const int*)d_in[3];    // [D]
  const float* qwt  = (const float*)d_in[4];  // [VOCAB]
  const float* dwt  = (const float*)d_in[5];  // [VOCAB]
  float* out = (float*)d_out;

  char* ws = (char*)d_ws;
  // Workspace layout (bytes):
  //   qhi: 1,048,576  qlo: 1,048,576  dhi: 8,388,608  dlo: 8,388,608
  //   pmax: 262,144   pidx: 262,144   total ~= 19.4 MB
  unsigned short* qhi = (unsigned short*)(ws + 0);
  unsigned short* qlo = (unsigned short*)(ws + 1048576);
  unsigned short* dhi = (unsigned short*)(ws + 2097152);
  unsigned short* dlo = (unsigned short*)(ws + 10485760);
  float*          pmax = (float*)(ws + 18874368);
  int*            pidx = (int*)  (ws + 19136512);

  // 1) fp32 -> bf16 hi/lo split
  {
    const int total = (QN + DN) * DIMN;        // 4,718,592
    const int blocks = (total + 255) / 256;
    convert_split_bf16<<<blocks, 256, 0, stream>>>(qe, de, qhi, qlo, dhi, dlo);
  }

  // 2) WMMA MaxSim: 32 q-groups x 16 d-slices; 8 waves/block share LDS doc tile
  maxsim_wmma<<<512, 256, 0, stream>>>(qhi, qlo, dhi, dlo, pmax, pidx);

  // 3) finalize scalar
  finalize_maxsim<<<1, 256, 0, stream>>>(pmax, pidx, qids, dids, qwt, dwt, out);
}
